// CrossAttention_62715112456471
// MI455X (gfx1250) — matrix operand order repaired and checked by reading, not compiled, not run on hardware
//
#include <hip/hip_runtime.h>

// ---------------------------------------------------------------------------
// MI455X (gfx1250) CrossAttention: all 5 GEMM stages + flash attention on
// v_wmma_f32_16x16x32_bf16 (wave32). GEMM operand tiles are staged into LDS
// with the Tensor Data Mover (tensor_load_to_lds, TENSORcnt, double-buffered),
// with TDM-inserted row padding for bank-conflict-free ds_load_b128 reads.
// ~77 GFLOP total, all intermediates resident in the 192MB L2.
// ---------------------------------------------------------------------------

typedef __bf16 bf16_t;
typedef bf16_t bfx4  __attribute__((ext_vector_type(4)));
typedef bf16_t bfx8  __attribute__((ext_vector_type(8)));
typedef bf16_t bfx16 __attribute__((ext_vector_type(16)));
typedef float  fx8   __attribute__((ext_vector_type(8)));
typedef unsigned int u32x4 __attribute__((ext_vector_type(4)));
typedef int i32x4 __attribute__((ext_vector_type(4)));
typedef int i32x8 __attribute__((ext_vector_type(8)));

__device__ __forceinline__ fx8 splat8(float v) {
  fx8 r;
#pragma unroll
  for (int i = 0; i < 8; ++i) r[i] = v;
  return r;
}

__device__ __forceinline__ fx8 wmma_bf16(bfx16 a, bfx16 b, fx8 c) {
  // (neg_a, A, neg_b, B, c_mod, C, reuse_a, reuse_b)
  return __builtin_amdgcn_wmma_f32_16x16x32_bf16(false, a, false, b, (short)0, c,
                                                 false, false);
}

// A-operand 16x32 bf16 (row-major source, ld in elements).
// ISA layout: lanes 0-15 = rows M=0..15; per lane elems 0..7 = K=half*8+0..7,
// elems 8..15 = K=16+half*8+0..7, where half = lane>>4.
__device__ __forceinline__ bfx16 load_frag_a(const bf16_t* a, int lda, int lane) {
  const int r = lane & 15;
  const int h = (lane >> 4) << 3;  // 0 or 8
  const bf16_t* p = a + (size_t)r * lda + h;
  bfx8 lo = *(const bfx8*)(p);
  bfx8 hi = *(const bfx8*)(p + 16);
  return __builtin_shufflevector(lo, hi, 0, 1, 2, 3, 4, 5, 6, 7, 8, 9, 10, 11,
                                 12, 13, 14, 15);
}

// B-operand 32x16 bf16 (row-major K x N source, ld in elements).
// ISA layout: lanes 0-15 = cols N=0..15; per lane elems 0..7 = K=half*8+0..7,
// elems 8..15 = K=16+half*8+0..7, where half = lane>>4.
__device__ __forceinline__ bfx16 load_frag_b(const bf16_t* b, int ldb, int lane) {
  const int n = lane & 15;
  const int h = (lane >> 4) << 3;  // 0 or 8
  bfx16 v;
#pragma unroll
  for (int s = 0; s < 8; ++s) {
    v[s] = b[(size_t)(h + s) * ldb + n];
    v[s + 8] = b[(size_t)(16 + h + s) * ldb + n];
  }
  return v;
}

// ---------------------------------------------------------------------------
// Tensor Data Mover: 2D tile load global -> LDS.
// Builds the D# per CDNA5 ISA:
//  group0: [1:0]=count=1, [63:32]=lds_addr, [120:64]=global_addr, type=2
//  group1: data_size=2B at [17:16], pad_enable [20], pad_interval [24:22],
//          pad_amount [31:25], tensor_dim0/1 = 2^30 (never OOB),
//          tile_dim0 [127:112], tile_dim1 [143:128], dim0 stride [207:160]
// Generic shared pointer -> LDS byte offset = low 32 address bits (flat LDS
// aperture truncation rule, ISA 00_overview §10.2).
// ---------------------------------------------------------------------------
__device__ __forceinline__ void tdm_load_2d(const void* lds_ptr,
                                            const void* gptr, uint32_t tile_d0,
                                            uint32_t tile_d1,
                                            uint32_t stride_elems,
                                            uint32_t pad_int_code,
                                            uint32_t pad_amt_code) {
  const uint64_t ga = (uint64_t)(uintptr_t)gptr;
  const uint32_t lds_off = (uint32_t)(uintptr_t)lds_ptr;
  u32x4 g0;
  g0[0] = 1u;  // count=1, user descriptor, no gather
  g0[1] = lds_off;
  g0[2] = (uint32_t)ga;
  g0[3] = (uint32_t)(ga >> 32) | (2u << 30);  // addr[56:32] | type=2
  i32x8 g1;
  g1[0] = (int)((1u << 16) |            // data_size = 2 bytes
                (1u << 20) |            // pad_enable
                (pad_int_code << 22) |  // pad interval (2^(c+1) DWORDs)
                (pad_amt_code << 25));  // pad amount (c+1 DWORDs)
  g1[1] = 0;                            // tensor_dim0[15:0]=0 (dim0=2^30)
  g1[2] = (int)0x4000u;                 // tensor_dim0[31:16] | dim1 lo = 0
  g1[3] = (int)(0x4000u | (tile_d0 << 16));  // tensor_dim1 hi | tile_dim0
  g1[4] = (int)tile_d1;                 // tile_dim1 | tile_dim2=0
  g1[5] = (int)stride_elems;            // tensor_dim0_stride[31:0]
  g1[6] = 0;
  g1[7] = 0;
  const i32x4 z4 = {0, 0, 0, 0};
#if __clang_major__ >= 23
  const i32x8 z8 = {0, 0, 0, 0, 0, 0, 0, 0};
  __builtin_amdgcn_tensor_load_to_lds(g0, g1, z4, z4, z8, 0);
#else
  __builtin_amdgcn_tensor_load_to_lds(g0, g1, z4, z4, 0);
#endif
}

// ---------------------------------------------------------------------------
// Generic bf16 GEMM: C[M,N] = A[M,K] * B[K,N], 256 threads = 8 waves,
// block tile 128x128, wave tile 32x64 (2x4 WMMA accumulators).
// TDM double-buffers A (128x32) and B (32x128) tiles into LDS; TDM padding
// gives row strides of 80B (A) / 272B (B): 16B aligned, spread across banks.
// EPI: 0 = store bf16, 1 = relu+store bf16, 2 = store fp32
// ---------------------------------------------------------------------------
#define LDA_T 40   // 32 + 8 pad elements
#define LDB_T 136  // 128 + 8 pad elements

template <int EPI>
__global__ __launch_bounds__(256) void gemm_bf16_kernel(
    const bf16_t* __restrict__ A, const bf16_t* __restrict__ B,
    void* __restrict__ Cout, int K, int lda, int ldb, int ldc) {
  __shared__ __align__(16) bf16_t ldsA[2][128 * LDA_T];
  __shared__ __align__(16) bf16_t ldsB[2][32 * LDB_T];
  const int lane = threadIdx.x & 31;
  const int wave = threadIdx.x >> 5;
  const int wm = (wave & 3) * 32;   // wave row offset in block tile
  const int wn = (wave >> 2) * 64;  // wave col offset in block tile
  const int bm = blockIdx.x * 128;
  const int bn = blockIdx.y * 128;

  fx8 acc[2][4];
#pragma unroll
  for (int i = 0; i < 2; ++i)
#pragma unroll
    for (int j = 0; j < 4; ++j) acc[i][j] = splat8(0.f);

  if (wave == 0) {
    // prologue: stage k0 = 0 tiles (TENSORcnt = 2)
    tdm_load_2d(ldsA[0], A + (size_t)bm * lda, 32, 128, (uint32_t)lda, 3, 3);
    tdm_load_2d(ldsB[0], B + bn, 128, 32, (uint32_t)ldb, 5, 3);
  }

  for (int k0 = 0, it = 0; k0 < K; k0 += 32, ++it) {
    const int buf = it & 1;
    if (wave == 0) {
      if (k0 + 32 < K) {  // stage next pair, then wait for current pair
        tdm_load_2d(ldsA[buf ^ 1], A + (size_t)bm * lda + (k0 + 32), 32, 128,
                    (uint32_t)lda, 3, 3);
        tdm_load_2d(ldsB[buf ^ 1], B + (size_t)(k0 + 32) * ldb + bn, 128, 32,
                    (uint32_t)ldb, 5, 3);
        __builtin_amdgcn_s_wait_tensorcnt(2);
      } else {
        __builtin_amdgcn_s_wait_tensorcnt(0);
      }
    }
    __syncthreads();  // current buffer visible to all waves

    const bf16_t* At = &ldsA[buf][wm * LDA_T];
    const bf16_t* Bt = &ldsB[buf][wn];
    bfx16 af[2];
    af[0] = load_frag_a(At, LDA_T, lane);
    af[1] = load_frag_a(At + 16 * LDA_T, LDA_T, lane);
#pragma unroll
    for (int j = 0; j < 4; ++j) {
      bfx16 bf = load_frag_b(Bt + j * 16, LDB_T, lane);
      acc[0][j] = wmma_bf16(af[0], bf, acc[0][j]);
      acc[1][j] = wmma_bf16(af[1], bf, acc[1][j]);
    }
    __syncthreads();  // all waves done with this buffer before TDM reuses it
  }

  // C/D layout: lane&15 = N col, VGPR slot r -> row = (lane>=16 ? 8:0)+r
  const int col0 = lane & 15;
  const int row_off = (lane >> 4) << 3;
#pragma unroll
  for (int i = 0; i < 2; ++i)
#pragma unroll
    for (int j = 0; j < 4; ++j)
#pragma unroll
      for (int r = 0; r < 8; ++r) {
        float v = acc[i][j][r];
        if (EPI == 1) v = fmaxf(v, 0.f);
        const size_t idx =
            (size_t)(bm + wm + i * 16 + row_off + r) * ldc + bn + wn + j * 16 +
            col0;
        if (EPI == 2)
          ((float*)Cout)[idx] = v;
        else
          ((bf16_t*)Cout)[idx] = (bf16_t)v;
      }
}

// ---------------------------------------------------------------------------
// fp32 -> bf16 conversion, 4 elems/thread (sizes are multiples of 1024)
// ---------------------------------------------------------------------------
__global__ __launch_bounds__(256) void cvt_f32_bf16_kernel(
    const float* __restrict__ in, bf16_t* __restrict__ out, int n4) {
  const int i = blockIdx.x * 256 + threadIdx.x;
  if (i < n4) {
    const float4 f = ((const float4*)in)[i];
    bfx4 r;
    r[0] = (bf16_t)f.x;
    r[1] = (bf16_t)f.y;
    r[2] = (bf16_t)f.z;
    r[3] = (bf16_t)f.w;
    ((bfx4*)out)[i] = r;
  }
}

// ---------------------------------------------------------------------------
// kT[b][c][j] = kv[b*2048+j][c]   (c in 0..1023: k half of the kv GEMM)
// makes q@k^T B-operand rows (fixed d, contiguous m) unit-stride.
// ---------------------------------------------------------------------------
__global__ __launch_bounds__(256) void build_kT_kernel(
    const bf16_t* __restrict__ kv, bf16_t* __restrict__ kT) {
  const int j = blockIdx.x * 256 + threadIdx.x;  // m index
  const int c = blockIdx.y;                      // 0..1023
  const int b = blockIdx.z;
  kT[((size_t)b * 1024 + c) * 2048 + j] = kv[((size_t)b * 2048 + j) * 2048 + c];
}

// ---------------------------------------------------------------------------
// Flash attention: grid (n/256, heads, batch), 8 waves/block, wave = 32 query
// rows. Stream m in chunks of 32: S=QK^T (WMMA), online softmax (cross-lane
// shfl reductions over the 16-lane row groups of the C layout), P staged bf16
// through per-wave LDS (C layout -> A layout), O += P*V (WMMA).
// ---------------------------------------------------------------------------
__global__ __launch_bounds__(256) void attn_kernel(
    const bf16_t* __restrict__ q,   // [b*2048, 1024] row-major
    const bf16_t* __restrict__ kT,  // [b][1024][2048]
    const bf16_t* __restrict__ kv,  // [b*2048, 2048]; v at col offset 1024
    bf16_t* __restrict__ out) {     // [b*2048, 1024]
  __shared__ __align__(16) bf16_t pshare[8][32 * 32];
  const int lane = threadIdx.x & 31;
  const int wave = threadIdx.x >> 5;
  const int b = blockIdx.z, h = blockIdx.y;
  const int n0 = blockIdx.x * 256 + wave * 32;

  const bf16_t* Q = q + (size_t)(b * 2048 + n0) * 1024 + h * 64;
  const bf16_t* KT = kT + (size_t)(b * 1024 + h * 64) * 2048;
  const bf16_t* V = kv + (size_t)b * 2048 * 2048 + 1024 + h * 64;
  const int col0 = lane & 15;
  const int row_off = (lane >> 4) << 3;
  const float scale = 0.125f;  // 1/sqrt(64)

  fx8 o[2][4], mst[2], lst[2];
#pragma unroll
  for (int i = 0; i < 2; ++i) {
#pragma unroll
    for (int j = 0; j < 4; ++j) o[i][j] = splat8(0.f);
    mst[i] = splat8(-3.0e38f);
    lst[i] = splat8(0.f);
  }

  // Q fragments held for the whole pass (K = d = 64 -> 2 k-steps, 2 row tiles)
  bfx16 qa[2][2];
#pragma unroll
  for (int i = 0; i < 2; ++i)
#pragma unroll
    for (int kk = 0; kk < 2; ++kk)
      qa[i][kk] = load_frag_a(Q + (size_t)i * 16 * 1024 + kk * 32, 1024, lane);

  bf16_t* P = pshare[wave];

  for (int m0 = 0; m0 < 2048; m0 += 32) {
    if (m0 + 32 < 2048) {  // global_prefetch_b8 of next chunk
      __builtin_prefetch(KT + m0 + 32, 0, 1);
      __builtin_prefetch(V + (size_t)(m0 + 32) * 2048, 0, 1);
    }
    // ---- S = scale * Q K^T  (2 row tiles x 2 col tiles) ----
    fx8 s[2][2];
#pragma unroll
    for (int i = 0; i < 2; ++i)
#pragma unroll
      for (int c = 0; c < 2; ++c) s[i][c] = splat8(0.f);
#pragma unroll
    for (int kk = 0; kk < 2; ++kk)
#pragma unroll
      for (int c = 0; c < 2; ++c) {
        bfx16 kb =
            load_frag_b(KT + (size_t)(kk * 32) * 2048 + m0 + c * 16, 2048, lane);
        s[0][c] = wmma_bf16(qa[0][kk], kb, s[0][c]);
        s[1][c] = wmma_bf16(qa[1][kk], kb, s[1][c]);
      }

    // ---- online softmax per row tile ----
#pragma unroll
    for (int i = 0; i < 2; ++i) {
      fx8 rm;
#pragma unroll
      for (int r = 0; r < 8; ++r) {
        s[i][0][r] *= scale;
        s[i][1][r] *= scale;
        rm[r] = fmaxf(s[i][0][r], s[i][1][r]);
      }
#pragma unroll
      for (int st = 0; st < 4; ++st) {  // reduce across the 16-lane row group
        const int mask = 1 << st;
#pragma unroll
        for (int r = 0; r < 8; ++r)
          rm[r] = fmaxf(rm[r], __shfl_xor(rm[r], mask, 32));
      }
      fx8 alpha;
#pragma unroll
      for (int r = 0; r < 8; ++r) {
        const float mnew = fmaxf(mst[i][r], rm[r]);
        alpha[r] = __expf(mst[i][r] - mnew);
        mst[i][r] = mnew;
        s[i][0][r] = __expf(s[i][0][r] - mnew);
        s[i][1][r] = __expf(s[i][1][r] - mnew);
      }
      fx8 rs;
#pragma unroll
      for (int r = 0; r < 8; ++r) rs[r] = s[i][0][r] + s[i][1][r];
#pragma unroll
      for (int st = 0; st < 4; ++st) {
        const int mask = 1 << st;
#pragma unroll
        for (int r = 0; r < 8; ++r) rs[r] += __shfl_xor(rs[r], mask, 32);
      }
#pragma unroll
      for (int r = 0; r < 8; ++r) lst[i][r] = lst[i][r] * alpha[r] + rs[r];
#pragma unroll
      for (int j = 0; j < 4; ++j)
#pragma unroll
        for (int r = 0; r < 8; ++r) o[i][j][r] *= alpha[r];
      // P tile (C layout) -> LDS row-major bf16 [32][32]
#pragma unroll
      for (int c = 0; c < 2; ++c)
#pragma unroll
        for (int r = 0; r < 8; ++r)
          P[(i * 16 + row_off + r) * 32 + c * 16 + col0] = (bf16_t)s[i][c][r];
    }

    // ---- O += P @ V  (K=32 chunk, 4 d-col tiles) ----
    bfx16 pa[2];
    pa[0] = load_frag_a(P, 32, lane);
    pa[1] = load_frag_a(P + 16 * 32, 32, lane);
#pragma unroll
    for (int j = 0; j < 4; ++j) {
      bfx16 vb = load_frag_b(V + (size_t)m0 * 2048 + j * 16, 2048, lane);
      o[0][j] = wmma_bf16(pa[0], vb, o[0][j]);
      o[1][j] = wmma_bf16(pa[1], vb, o[1][j]);
    }
  }

  // ---- normalize + store bf16 ----
#pragma unroll
  for (int i = 0; i < 2; ++i) {
    fx8 inv;
#pragma unroll
    for (int r = 0; r < 8; ++r) inv[r] = 1.f / lst[i][r];
#pragma unroll
    for (int j = 0; j < 4; ++j)
#pragma unroll
      for (int r = 0; r < 8; ++r)
        out[(size_t)(b * 2048 + n0 + i * 16 + row_off + r) * 1024 + h * 64 +
            j * 16 + col0] = (bf16_t)(o[i][j][r] * inv[r]);
  }
}

// ---------------------------------------------------------------------------
// In-place LayerNorm over last dim (1024) with gamma; one block per row.
// ---------------------------------------------------------------------------
__global__ __launch_bounds__(256) void layernorm_kernel(
    float* __restrict__ y, const float* __restrict__ gamma) {
  float* p = y + (size_t)blockIdx.x * 1024;
  const int t = threadIdx.x;
  float v[4], s = 0.f, s2 = 0.f;
#pragma unroll
  for (int i = 0; i < 4; ++i) {
    v[i] = p[t + i * 256];
    s += v[i];
    s2 += v[i] * v[i];
  }
#pragma unroll
  for (int st = 0; st < 5; ++st) {
    const int mask = 1 << st;
    s += __shfl_xor(s, mask, 32);
    s2 += __shfl_xor(s2, mask, 32);
  }
  __shared__ float rs[8], rs2[8];
  const int wave = t >> 5, lane = t & 31;
  if (lane == 0) {
    rs[wave] = s;
    rs2[wave] = s2;
  }
  __syncthreads();
  s = 0.f;
  s2 = 0.f;
#pragma unroll
  for (int w = 0; w < 8; ++w) {
    s += rs[w];
    s2 += rs2[w];
  }
  const float mean = s * (1.f / 1024.f);
  const float var = s2 * (1.f / 1024.f) - mean * mean;
  const float rstd = rsqrtf(var + 1e-5f);
#pragma unroll
  for (int i = 0; i < 4; ++i) {
    const int c = t + i * 256;
    p[c] = (v[i] - mean) * rstd * gamma[c];
  }
}

// ---------------------------------------------------------------------------
// Host-side orchestration
// ---------------------------------------------------------------------------
extern "C" void kernel_launch(void* const* d_in, const int* in_sizes, int n_in,
                              void* d_out, int out_size, void* d_ws,
                              size_t ws_size, hipStream_t stream) {
  (void)in_sizes; (void)n_in; (void)out_size; (void)ws_size;
  const float* x = (const float*)d_in[0];       // [2,2048,1024]
  const float* ctx = (const float*)d_in[1];     // [2,2048,1024]
  const float* w_kv = (const float*)d_in[2];    // [1024,2048]
  const float* w_q = (const float*)d_in[3];     // [1024,1024]
  const float* w1 = (const float*)d_in[4];      // [1024,1024]
  const float* w2 = (const float*)d_in[5];      // [1024,1024]
  const float* gamma = (const float*)d_in[6];   // [1024]
  float* out = (float*)d_out;                   // [2,2048,1024] fp32

  // workspace carve-up (~74 MB, all 256B aligned)
  char* ws = (char*)d_ws;
  size_t off = 0;
  auto carve = [&](size_t bytes) {
    void* p = ws + off;
    off += (bytes + 255) & ~(size_t)255;
    return p;
  };
  bf16_t* xb    = (bf16_t*)carve((size_t)4096 * 1024 * 2);
  bf16_t* ctxb  = (bf16_t*)carve((size_t)4096 * 1024 * 2);
  bf16_t* wkvb  = (bf16_t*)carve((size_t)1024 * 2048 * 2);
  bf16_t* wqb   = (bf16_t*)carve((size_t)1024 * 1024 * 2);
  bf16_t* w1b   = (bf16_t*)carve((size_t)1024 * 1024 * 2);
  bf16_t* w2b   = (bf16_t*)carve((size_t)1024 * 1024 * 2);
  bf16_t* kvb   = (bf16_t*)carve((size_t)4096 * 2048 * 2);
  bf16_t* kTb   = (bf16_t*)carve((size_t)2 * 1024 * 2048 * 2);
  bf16_t* qb    = (bf16_t*)carve((size_t)4096 * 1024 * 2);
  bf16_t* attnb = (bf16_t*)carve((size_t)4096 * 1024 * 2);
  bf16_t* hidb  = (bf16_t*)carve((size_t)4096 * 1024 * 2);

  // 1) fp32 -> bf16 conversions
  cvt_f32_bf16_kernel<<<4096, 256, 0, stream>>>(x, xb, 1048576);
  cvt_f32_bf16_kernel<<<4096, 256, 0, stream>>>(ctx, ctxb, 1048576);
  cvt_f32_bf16_kernel<<<2048, 256, 0, stream>>>(w_kv, wkvb, 524288);
  cvt_f32_bf16_kernel<<<1024, 256, 0, stream>>>(w_q, wqb, 262144);
  cvt_f32_bf16_kernel<<<1024, 256, 0, stream>>>(w1, w1b, 262144);
  cvt_f32_bf16_kernel<<<1024, 256, 0, stream>>>(w2, w2b, 262144);

  // 2) kv = context @ w_kv   [4096,2048]
  gemm_bf16_kernel<0><<<dim3(32, 16), 256, 0, stream>>>(ctxb, wkvb, kvb, 1024,
                                                        1024, 2048, 2048);
  // 3) q = x @ w_q           [4096,1024]
  gemm_bf16_kernel<0><<<dim3(32, 8), 256, 0, stream>>>(xb, wqb, qb, 1024, 1024,
                                                       1024, 1024);
  // 4) kT[b][d][m]
  build_kT_kernel<<<dim3(8, 1024, 2), 256, 0, stream>>>(kvb, kTb);

  // 5) attention -> attnb [4096,1024] bf16
  attn_kernel<<<dim3(8, 16, 2), 256, 0, stream>>>(qb, kTb, kvb, attnb);

  // 6) hidden = relu(attn @ w_out1)
  gemm_bf16_kernel<1><<<dim3(32, 8), 256, 0, stream>>>(attnb, w1b, hidb, 1024,
                                                       1024, 1024, 1024);
  // 7) y = hidden @ w_out2 -> fp32 d_out
  gemm_bf16_kernel<2><<<dim3(32, 8), 256, 0, stream>>>(hidb, w2b, out, 1024,
                                                       1024, 1024, 1024);
  // 8) layernorm(gamma) in place on d_out
  layernorm_kernel<<<4096, 256, 0, stream>>>(out, gamma);
}
